// VectorQuantizerMultiHead_11725260718117
// MI455X (gfx1250) — matrix operand order repaired and checked by reading, not compiled, not run on hardware
//
#include <hip/hip_runtime.h>

typedef __attribute__((ext_vector_type(16))) __bf16 v16bf;
typedef __attribute__((ext_vector_type(8)))  __bf16 v8bf;
typedef __attribute__((ext_vector_type(8)))  float  v8f;
typedef __attribute__((ext_vector_type(4)))  int    v4i;
typedef __attribute__((address_space(1))) v4i* v4i_gptr;   // global int4*
typedef __attribute__((address_space(3))) v4i* v4i_lptr;   // LDS int4*

#define B_N  4096
#define H_N  4
#define K_N  8192
#define DH_N 256
#define D_N  1024
#define MT   64      // rows per block
#define KC   64      // codes per K-chunk
#define NCH  (K_N / KC)

// padded LDS row strides (elements), 16B-aligned rows except S_s (260 B)
#define XS_LD  264
#define CBT_LD 72
#define CB_LD  264
#define S_LD   65
#define P_LD   72

// ---- CDNA5 async global->LDS path (hedged: falls back to sync copy) ----
#if defined(__gfx1250__) && __has_builtin(__builtin_amdgcn_global_load_async_to_lds_b128)
#define HAVE_ASYNC 1
#define ASYNC_CP16(dst, src)                                                   \
  __builtin_amdgcn_global_load_async_to_lds_b128(                              \
      (v4i_gptr)(src), (v4i_lptr)(dst), 0, 0)
#else
#define HAVE_ASYNC 0
#define ASYNC_CP16(dst, src) (*(uint4*)(dst) = *(const uint4*)(src))
#endif

__device__ __forceinline__ void wait_async_prev16() {
#if HAVE_ASYNC
#if __has_builtin(__builtin_amdgcn_s_wait_asynccnt)
  __builtin_amdgcn_s_wait_asynccnt(16);
#else
  asm volatile("s_wait_asynccnt 0x10" ::: "memory");
#endif
#endif
}
__device__ __forceinline__ void wait_async_all() {
#if HAVE_ASYNC
#if __has_builtin(__builtin_amdgcn_s_wait_asynccnt)
  __builtin_amdgcn_s_wait_asynccnt(0);
#else
  asm volatile("s_wait_asynccnt 0x0" ::: "memory");
#endif
#endif
}

__device__ __forceinline__ unsigned int pk2bf(float a, float b) {
  union { float f; unsigned int u; } ca, cb;
  ca.f = a; cb.f = b;
  unsigned int ua = ca.u + 0x7FFFu + ((ca.u >> 16) & 1u); // RNE f32->bf16
  unsigned int ub = cb.u + 0x7FFFu + ((cb.u >> 16) & 1u);
  return (ua >> 16) | (ub & 0xFFFF0000u);
}

// ---- prep 1: codebook f32 -> bf16, per-code squared norm; zero the loss slot
__global__ __launch_bounds__(256) void prep_convert(const float* __restrict__ cb,
                                                    __bf16* __restrict__ cb_bf,
                                                    float* __restrict__ wsq,
                                                    float* __restrict__ out_loss) {
  const int row  = blockIdx.x * 8 + (threadIdx.x >> 5);  // H*K rows
  const int lane = threadIdx.x & 31;
  const float* src = cb + (size_t)row * DH_N + lane * 8;
  float4 v0 = ((const float4*)src)[0];
  float4 v1 = ((const float4*)src)[1];
  uint4 p = make_uint4(pk2bf(v0.x, v0.y), pk2bf(v0.z, v0.w),
                       pk2bf(v1.x, v1.y), pk2bf(v1.z, v1.w));
  *(uint4*)(cb_bf + (size_t)row * DH_N + lane * 8) = p;
  float s = v0.x*v0.x + v0.y*v0.y + v0.z*v0.z + v0.w*v0.w
          + v1.x*v1.x + v1.y*v1.y + v1.z*v1.z + v1.w*v1.w;
  #pragma unroll
  for (int off = 16; off >= 1; off >>= 1) s += __shfl_xor(s, off);
  if (lane == 0) wsq[row] = s;
  if (blockIdx.x == 0 && threadIdx.x == 0) *out_loss = 0.f;
}

// ---- prep 2: build transposed bf16 codebook [H, DH, K] via 64x64 LDS tiles
__global__ __launch_bounds__(256) void prep_transpose(const __bf16* __restrict__ cb_bf,
                                                      __bf16* __restrict__ cbT) {
  __shared__ __align__(16) __bf16 tile[64][CBT_LD];
  const int bid = blockIdx.x;
  const int h  = bid >> 9;          // 512 blocks per head
  const int kt = (bid & 511) >> 2;  // 0..127
  const int dt = bid & 3;           // 0..3
  const int t = threadIdx.x;
  #pragma unroll
  for (int p = 0; p < 2; ++p) {
    int idx = t + p * 256;
    int k = idx >> 3, du = idx & 7;
    const uint4* src = (const uint4*)(cb_bf +
        (size_t)(h * K_N + kt * 64 + k) * DH_N + dt * 64 + du * 8);
    *(uint4*)&tile[k][du * 8] = *src;
  }
  __syncthreads();
  #pragma unroll
  for (int p = 0; p < 2; ++p) {
    int idx = t + p * 256;
    int d = idx >> 3, ku = idx & 7;
    __align__(16) __bf16 tmp[8];
    #pragma unroll
    for (int i = 0; i < 8; ++i) tmp[i] = tile[ku * 8 + i][d];
    *(uint4*)(cbT + (size_t)(h * DH_N + dt * 64 + d) * K_N + kt * 64 + ku * 8) =
        *(const uint4*)tmp;
  }
}

// ---- fused VQ: scores GEMM -> online softmax (+argmax) -> P@codebook GEMM -> loss
__global__ __launch_bounds__(256) void vq_fused(const float* __restrict__ x,     // [B, D]
                                                const __bf16* __restrict__ cb,   // [H*K, DH]
                                                const __bf16* __restrict__ cbT,  // [H*DH, K]
                                                const float* __restrict__ wsq,   // [H*K]
                                                float* __restrict__ out_q,       // [B, D]
                                                int* __restrict__ out_codes,     // [B, H]
                                                float* __restrict__ out_loss) {
  __shared__ __align__(16) __bf16 xs[MT][XS_LD];
  __shared__ __align__(16) __bf16 cbt_s[2][DH_N][CBT_LD];   // double-buffered
  __shared__ __align__(16) __bf16 cb_s[2][KC][CB_LD];       // double-buffered
  __shared__ __align__(16) float  S_s[MT][S_LD];
  __shared__ __align__(16) __bf16 P_s[MT][P_LD];
  __shared__ float wsq_s[KC];
  __shared__ float m_s[MT], l_s[MT], sc_s[MT], bv_s[MT];
  __shared__ int   bi_s[MT];
  __shared__ float red_s[8];

  const int t    = threadIdx.x;
  const int h    = blockIdx.x & (H_N - 1);
  const int row0 = (blockIdx.x >> 2) * MT;
  const int wave = t >> 5;
  const int lane = t & 31;
  const int rg   = wave >> 1;          // row group 0..3 (16 rows each)
  const int sub  = wave & 1;           // k-half (GEMM1) / d-half (GEMM2)
  const bool hi  = lane >= 16;
  const int l16  = lane & 15;

  const __bf16* cbT_h = cbT + (size_t)h * DH_N * K_N;
  const __bf16* cb_h  = cb  + (size_t)h * K_N * DH_N;
  const float*  wsq_h = wsq + h * K_N;

  // per-thread staging coordinates (16B granules)
  const int cr = t >> 2, cq = t & 3;   // cb chunk: row cr (64 rows), quarter cq

  // stage x tile [64][256] f32 -> bf16 LDS
  {
    const float* src = x + (size_t)(row0 + cr) * D_N + h * DH_N + cq * 64;
    #pragma unroll
    for (int i = 0; i < 16; ++i) {
      float4 v = ((const float4*)src)[i];
      *(uint2*)&xs[cr][cq * 64 + i * 4] =
          make_uint2(pk2bf(v.x, v.y), pk2bf(v.z, v.w));
    }
  }
  if (t < MT) { m_s[t] = -__builtin_inff(); l_s[t] = 0.f; bv_s[t] = -__builtin_inff(); bi_s[t] = 0; }

  v8f acc[8];
  v8f vzero = {};
  #pragma unroll
  for (int f = 0; f < 8; ++f) acc[f] = vzero;

  // prologue: async-prefetch chunk 0 into buffer 0 (16 x b128 per wave)
  {
    const __bf16* g1 = cbT_h + (size_t)t * K_N;      // kc = 0
    #pragma unroll
    for (int i = 0; i < 8; ++i) ASYNC_CP16(&cbt_s[0][t][i * 8], g1 + i * 8);
    const __bf16* g2 = cb_h + (size_t)cr * DH_N + cq * 64;
    #pragma unroll
    for (int i = 0; i < 8; ++i) ASYNC_CP16(&cb_s[0][cr][cq * 64 + i * 8], g2 + i * 8);
  }

  for (int ic = 0; ic < NCH; ++ic) {
    const int p  = ic & 1;
    const int kc = ic * KC;

    // prefetch next chunk into the other buffer (freed by last iter's end barrier)
    if (ic + 1 < NCH) {
      const int kn = kc + KC;
      const __bf16* g1 = cbT_h + (size_t)t * K_N + kn;
      #pragma unroll
      for (int i = 0; i < 8; ++i) ASYNC_CP16(&cbt_s[1 - p][t][i * 8], g1 + i * 8);
      const __bf16* g2 = cb_h + (size_t)(kn + cr) * DH_N + cq * 64;
      #pragma unroll
      for (int i = 0; i < 8; ++i) ASYNC_CP16(&cb_s[1 - p][cr][cq * 64 + i * 8], g2 + i * 8);
      wait_async_prev16();   // current chunk's 16 done; next chunk's 16 in flight
    } else {
      wait_async_all();
    }
    if (t < KC) wsq_s[t] = wsq_h[kc + t];
    __syncthreads();

    // GEMM1: wave(rg,sub) -> scores [16 rows][32 k], k base sub*32
    v8f s0 = {}, s1 = {};
    #pragma unroll
    for (int d0 = 0; d0 < DH_N; d0 += 32) {
      v16bf a;
      const __bf16* ap = &xs[rg * 16 + l16][d0 + (hi ? 8 : 0)];
      ((v8bf*)&a)[0] = *(const v8bf*)ap;
      ((v8bf*)&a)[1] = *(const v8bf*)(ap + 16);
      const __bf16* bp = &cb_s[p][sub * 32 + l16][d0 + (hi ? 8 : 0)];
      v16bf b0, b1;
      ((v8bf*)&b0)[0] = *(const v8bf*)bp;
      ((v8bf*)&b0)[1] = *(const v8bf*)(bp + 16);
      ((v8bf*)&b1)[0] = *(const v8bf*)(bp + 16 * CB_LD);
      ((v8bf*)&b1)[1] = *(const v8bf*)(bp + 16 * CB_LD + 16);
      s0 = __builtin_amdgcn_wmma_f32_16x16x32_bf16(false, a, false, b0, (short)0, s0, false, false);
      s1 = __builtin_amdgcn_wmma_f32_16x16x32_bf16(false, a, false, b1, (short)0, s1, false, false);
    }
    #pragma unroll
    for (int j = 0; j < 8; ++j) {
      int rr = rg * 16 + j + (hi ? 8 : 0);
      S_s[rr][sub * 32 + l16]      = s0[j];
      S_s[rr][sub * 32 + 16 + l16] = s1[j];
    }
    __syncthreads();

    // online softmax + argmax: quad of threads per row
    {
      float sv[16];
      float mloc = -__builtin_inff(); int iloc = 0;
      #pragma unroll
      for (int i = 0; i < 16; ++i) {
        int k = cq * 16 + i;
        float s = 2.f * S_s[cr][k] - wsq_s[k];
        sv[i] = s;
        if (s > mloc) { mloc = s; iloc = kc + k; }
      }
      #pragma unroll
      for (int off = 1; off <= 2; off <<= 1) {
        float om = __shfl_xor(mloc, off);
        int   oi = __shfl_xor(iloc, off);
        if (om > mloc || (om == mloc && oi < iloc)) { mloc = om; iloc = oi; }
      }
      float mold = m_s[cr];
      float mnew = fmaxf(mold, mloc);
      float scale = __expf(mold - mnew);
      float psum = 0.f;
      #pragma unroll
      for (int i = 0; i < 16; i += 2) {
        float p0 = __expf(sv[i]     - mnew);
        float p1 = __expf(sv[i + 1] - mnew);
        psum += p0 + p1;
        *(unsigned int*)&P_s[cr][cq * 16 + i] = pk2bf(p0, p1);
      }
      psum += __shfl_xor(psum, 1);
      psum += __shfl_xor(psum, 2);
      if (cq == 0) {
        l_s[cr] = l_s[cr] * scale + psum;
        m_s[cr] = mnew;
        sc_s[cr] = scale;
        if (mloc > bv_s[cr]) { bv_s[cr] = mloc; bi_s[cr] = iloc; }
      }
    }
    __syncthreads();

    // rescale accumulators by exp(m_old - m_new) per row
    float scj[8];
    #pragma unroll
    for (int j = 0; j < 8; ++j) scj[j] = sc_s[rg * 16 + j + (hi ? 8 : 0)];
    #pragma unroll
    for (int f = 0; f < 8; ++f)
      #pragma unroll
      for (int j = 0; j < 8; ++j) acc[f][j] *= scj[j];

    // GEMM2: wave(rg,sub=dg): acc[16 rows][128 d] += P[16][64] x cb[64][128]
    #pragma unroll
    for (int kcc = 0; kcc < 2; ++kcc) {
      v16bf a;
      const __bf16* ap = &P_s[rg * 16 + l16][kcc * 32 + (hi ? 8 : 0)];
      ((v8bf*)&a)[0] = *(const v8bf*)ap;
      ((v8bf*)&a)[1] = *(const v8bf*)(ap + 16);
      int krow = kcc * 32 + lane;
      #pragma unroll
      for (int f = 0; f < 8; ++f) {
        const __bf16* bp = &cbt_s[p][(sub * 8 + f) * 16 + l16][kcc * 32 + (hi ? 8 : 0)];
        v16bf b;
        ((v8bf*)&b)[0] = *(const v8bf*)bp;
        ((v8bf*)&b)[1] = *(const v8bf*)(bp + 16);
        acc[f] = __builtin_amdgcn_wmma_f32_16x16x32_bf16(false, a, false, b, (short)0, acc[f], false, false);
      }
    }
    __syncthreads();   // end-of-compute: frees buffer p for next iter's prefetch
  }

  // epilogue: normalize, write quantized, fused loss partial
  float inv[8];
  #pragma unroll
  for (int j = 0; j < 8; ++j) inv[j] = 1.f / l_s[rg * 16 + j + (hi ? 8 : 0)];
  float lsum = 0.f;
  #pragma unroll
  for (int f = 0; f < 8; ++f) {
    int dcol = sub * 128 + f * 16 + l16;
    #pragma unroll
    for (int j = 0; j < 8; ++j) {
      int rr = row0 + rg * 16 + j + (hi ? 8 : 0);
      float v = acc[f][j] * inv[j];
      size_t oi = (size_t)rr * D_N + h * DH_N + dcol;
      out_q[oi] = v;
      float dd = v - x[oi];
      lsum += dd * dd;
    }
  }
  #pragma unroll
  for (int off = 16; off >= 1; off >>= 1) lsum += __shfl_xor(lsum, off);
  if (lane == 0) red_s[wave] = lsum;
  __syncthreads();
  if (t == 0) {
    float tot = 0.f;
    #pragma unroll
    for (int i = 0; i < 8; ++i) tot += red_s[i];
    atomicAdd(out_loss, tot * (1.25f / (float)(B_N * D_N)));
  }
  if (t < MT) out_codes[(size_t)(row0 + t) * H_N + h] = bi_s[t];
}

extern "C" void kernel_launch(void* const* d_in, const int* in_sizes, int n_in,
                              void* d_out, int out_size, void* d_ws, size_t ws_size,
                              hipStream_t stream) {
  const float* x  = (const float*)d_in[0];   // inputs [B,1,D]
  const float* cb = (const float*)d_in[1];   // codebook [H,K,DH]
  char* ws = (char*)d_ws;
  const size_t cb_bytes = (size_t)H_N * K_N * DH_N * 2;    // 16.8 MB
  __bf16* cb_bf  = (__bf16*)ws;
  __bf16* cbT_bf = (__bf16*)(ws + cb_bytes);
  float*  wsq    = (float*)(ws + 2 * cb_bytes);

  float* out      = (float*)d_out;
  float* out_loss = out;                       // [0]
  float* out_q    = out + 1;                   // [B*D]
  int*   out_codes = (int*)(out + 1 + (size_t)B_N * D_N);  // [B*H]

  prep_convert<<<H_N * K_N / 8, 256, 0, stream>>>(cb, cb_bf, wsq, out_loss);
  prep_transpose<<<H_N * (K_N / 64) * (DH_N / 64), 256, 0, stream>>>(cb_bf, cbT_bf);
  vq_fused<<<H_N * (B_N / MT), 256, 0, stream>>>(x, cb_bf, cbT_bf, wsq,
                                                 out_q, out_codes, out_loss);
}
